// MaskNet_84138409328693
// MI455X (gfx1250) — matrix operand order repaired and checked by reading, not compiled, not run on hardware
//
#include <hip/hip_runtime.h>
#include <hip/hip_bf16.h>
#include <math.h>

typedef __bf16 bf16_t;
typedef __bf16 v16bf __attribute__((ext_vector_type(16)));
typedef __bf16 v8bf  __attribute__((ext_vector_type(8)));
typedef float  v8f   __attribute__((ext_vector_type(8)));

constexpr int NB      = 1024;           // batch
constexpr int NF      = 64;             // features
constexpr int ND      = 256;            // D
constexpr int NM      = 32;             // M
constexpr int KTOP    = ND / 3;         // 85
constexpr int IN_DIM  = NF * 2 * ND;    // 32768
constexpr int OUT_DIM = NM * ND;        // 8192
constexpr int KSPLIT  = 8;              // split-K factor for GEMM1
constexpr float LN_EPS = 1e-3f;

// ---------------------------------------------------------------------------
// Pack x = flatten(concat([mean, var], -1)) into bf16, row-major (NB, IN_DIM).
// ---------------------------------------------------------------------------
__global__ void pack_x_bf16(const float* __restrict__ mean,
                            const float* __restrict__ varr,
                            bf16_t* __restrict__ x) {
    int i  = blockIdx.x * blockDim.x + threadIdx.x;  // over NB*NF*ND
    int d  = i % ND;
    int bf = i / ND;                                 // b*NF + f
    long base = (long)bf * (2 * ND) + d;
    x[base]      = (bf16_t)mean[i];
    x[base + ND] = (bf16_t)varr[i];
}

// ---------------------------------------------------------------------------
// LDS-tiled fp32 -> bf16 transpose: out[c][r] = (bf16) in[r][c]
// ---------------------------------------------------------------------------
__global__ void transpose_bf16(const float* __restrict__ in,
                               bf16_t* __restrict__ out,
                               int rows, int cols) {
    __shared__ float tile[32][33];
    int c0 = blockIdx.x * 32;
    int r0 = blockIdx.y * 32;
    for (int j = threadIdx.y; j < 32; j += 8)
        tile[j][threadIdx.x] = in[(long)(r0 + j) * cols + (c0 + threadIdx.x)];
    __syncthreads();
    for (int j = threadIdx.y; j < 32; j += 8)
        out[(long)(c0 + j) * rows + (r0 + threadIdx.x)] = (bf16_t)tile[threadIdx.x][j];
}

// ---------------------------------------------------------------------------
// WMMA bf16 GEMM, ping-pong double-buffered fragments (no rotation copies).
//   A: MxK row-major bf16, Bt: NxK row-major bf16 (B transposed).
// Wave tile 16(M)x64(N), block = 4 waves stacked in M -> 64x64 tile.
// gridDim.z = K-split; kChunk must be a multiple of 64.
//   mode 0: raw f32 partial into C + z*M*N    mode 1: fused bias + tanh
// ---------------------------------------------------------------------------
__device__ __forceinline__ v16bf load_frag16(const bf16_t* p0, const bf16_t* p1) {
    v8bf lo = *(const v8bf*)p0;     // 16B aligned
    v8bf hi = *(const v8bf*)p1;
    v16bf r;
#pragma unroll
    for (int i = 0; i < 8; ++i) { r[i] = lo[i]; r[i + 8] = hi[i]; }
    return r;
}

__global__ __launch_bounds__(128)
void wmma_gemm_bf16(const bf16_t* __restrict__ A,
                    const bf16_t* __restrict__ Bt,
                    const float* __restrict__ bias,
                    float* __restrict__ C,
                    int M, int N, int Kd, int kChunk, int mode) {
    const int lane = threadIdx.x & 31;
    const int wave = threadIdx.x >> 5;
    const int half = lane >> 4;       // 0: lanes 0-15, 1: lanes 16-31
    const int l16  = lane & 15;
    const int mBase = blockIdx.y * 64 + wave * 16;
    const int nBase = blockIdx.x * 64;
    const int kStart = blockIdx.z * kChunk;
    const int kEnd   = kStart + kChunk;

    // A fragment (16x32 bf16): lanes 0-15 hold K 0-15 of row l16,
    // lanes 16-31 hold K 16-31 (contiguous, matching B).
    const bf16_t* arow = A + (long)(mBase + l16) * Kd + half * 16;
    // B fragment (32x16 bf16): lane holds one column, 16 contiguous K values.
    const bf16_t* brow[4];
#pragma unroll
    for (int t = 0; t < 4; ++t)
        brow[t] = Bt + (long)(nBase + t * 16 + l16) * Kd + half * 16;

    v8f acc[4] = {v8f{}, v8f{}, v8f{}, v8f{}};

    // ---- prologue: buffer A <- k-step kStart --------------------------------
    v16bf afA = load_frag16(arow + kStart, arow + kStart + 8);
    v16bf bfA[4];
#pragma unroll
    for (int t = 0; t < 4; ++t)
        bfA[t] = load_frag16(brow[t] + kStart, brow[t] + kStart + 8);

    v16bf afB;
    v16bf bfB[4];

    // ---- unroll-by-2 ping-pong: every frag reg is defined by its load -------
    int k = kStart;
    for (; k < kEnd - 64; k += 64) {
        // load buffer B <- step k+32
        __builtin_prefetch(arow + k + 1024, 0, 1);   // global_prefetch_b8
        afB = load_frag16(arow + k + 32, arow + k + 40);
#pragma unroll
        for (int t = 0; t < 4; ++t)
            bfB[t] = load_frag16(brow[t] + k + 32, brow[t] + k + 40);
        // compute step k from buffer A
#pragma unroll
        for (int t = 0; t < 4; ++t)
            acc[t] = __builtin_amdgcn_wmma_f32_16x16x32_bf16(
                false, afA, false, bfA[t], (short)0, acc[t], false, false);
        // load buffer A <- step k+64
        afA = load_frag16(arow + k + 64, arow + k + 72);
#pragma unroll
        for (int t = 0; t < 4; ++t)
            bfA[t] = load_frag16(brow[t] + k + 64, brow[t] + k + 72);
        // compute step k+32 from buffer B
#pragma unroll
        for (int t = 0; t < 4; ++t)
            acc[t] = __builtin_amdgcn_wmma_f32_16x16x32_bf16(
                false, afB, false, bfB[t], (short)0, acc[t], false, false);
    }

    // ---- tail: exactly two k-steps remain (kChunk % 64 == 0) ----------------
    afB = load_frag16(arow + k + 32, arow + k + 40);
#pragma unroll
    for (int t = 0; t < 4; ++t)
        bfB[t] = load_frag16(brow[t] + k + 32, brow[t] + k + 40);
#pragma unroll
    for (int t = 0; t < 4; ++t)
        acc[t] = __builtin_amdgcn_wmma_f32_16x16x32_bf16(
            false, afA, false, bfA[t], (short)0, acc[t], false, false);
#pragma unroll
    for (int t = 0; t < 4; ++t)
        acc[t] = __builtin_amdgcn_wmma_f32_16x16x32_bf16(
            false, afB, false, bfB[t], (short)0, acc[t], false, false);

    // ---- store: C/D layout -> vgpr v = row (v + 8*half), col l16 ------------
    if (mode == 0) {
        float* Cp = C + (long)blockIdx.z * M * N;
#pragma unroll
        for (int t = 0; t < 4; ++t) {
            int n = nBase + t * 16 + l16;
#pragma unroll
            for (int v = 0; v < 8; ++v) {
                int m = mBase + v + half * 8;
                Cp[(long)m * N + n] = acc[t][v];
            }
        }
    } else {
#pragma unroll
        for (int t = 0; t < 4; ++t) {
            int n = nBase + t * 16 + l16;
            float bv = bias[n];
#pragma unroll
            for (int v = 0; v < 8; ++v) {
                int m = mBase + v + half * 8;
                C[(long)m * N + n] = tanhf(acc[t][v] + bv);
            }
        }
    }
}

// ---------------------------------------------------------------------------
// Split-K reduction (fixed order -> deterministic) + bias + exact GELU.
// ---------------------------------------------------------------------------
__global__ void reduce_bias_gelu(const float* __restrict__ P,
                                 const float* __restrict__ bias,
                                 float* __restrict__ H) {
    int i = blockIdx.x * blockDim.x + threadIdx.x;   // over NB*ND
    float s = 0.0f;
#pragma unroll
    for (int p = 0; p < KSPLIT; ++p)
        s += P[(long)p * NB * ND + i];
    float v = s + bias[i % ND];
    H[i] = 0.5f * v * (1.0f + erff(v * 0.70710678118654752f));
}

// ---------------------------------------------------------------------------
// LayerNorm over D=256 + exact kWTA (top-85 by |h|, index tie-break).
// ---------------------------------------------------------------------------
__global__ void ln_kwta(const float* __restrict__ H,
                        const float* __restrict__ gamma,
                        const float* __restrict__ beta,
                        bf16_t* __restrict__ Hbf) {
    __shared__ float red[ND];
    __shared__ float sab[ND];
    const int row = blockIdx.x;
    const int t   = threadIdx.x;

    float h = H[row * ND + t];

    red[t] = h;
    __syncthreads();
    for (int s = ND / 2; s > 0; s >>= 1) {
        if (t < s) red[t] += red[t + s];
        __syncthreads();
    }
    float mu = red[0] * (1.0f / ND);
    __syncthreads();

    float dif = h - mu;
    red[t] = dif * dif;
    __syncthreads();
    for (int s = ND / 2; s > 0; s >>= 1) {
        if (t < s) red[t] += red[t + s];
        __syncthreads();
    }
    float var = red[0] * (1.0f / ND);

    float hn = dif * rsqrtf(var + LN_EPS) * gamma[t] + beta[t];
    sab[t] = fabsf(hn);
    __syncthreads();

    float a = sab[t];
    int rank = 0;
#pragma unroll 8
    for (int j = 0; j < ND; ++j) {
        float b = sab[j];
        rank += (b > a) || (b == a && j < t);
    }
    Hbf[row * ND + t] = (bf16_t)(rank < KTOP ? hn : 0.0f);
}

// ---------------------------------------------------------------------------
extern "C" void kernel_launch(void* const* d_in, const int* in_sizes, int n_in,
                              void* d_out, int out_size, void* d_ws, size_t ws_size,
                              hipStream_t stream) {
    const float* mean  = (const float*)d_in[0];
    const float* varr  = (const float*)d_in[1];
    const float* W1    = (const float*)d_in[2];
    const float* b1    = (const float*)d_in[3];
    const float* gamma = (const float*)d_in[4];
    const float* beta  = (const float*)d_in[5];
    const float* W2    = (const float*)d_in[6];
    const float* b2    = (const float*)d_in[7];
    float* out = (float*)d_out;

    char* ws = (char*)d_ws;
    bf16_t* Xbf  = (bf16_t*)ws;  ws += (size_t)NB * IN_DIM * sizeof(bf16_t);
    bf16_t* W1T  = (bf16_t*)ws;  ws += (size_t)IN_DIM * ND * sizeof(bf16_t);
    bf16_t* W2T  = (bf16_t*)ws;  ws += (size_t)ND * OUT_DIM * sizeof(bf16_t);
    float*  Part = (float*)ws;   ws += (size_t)KSPLIT * NB * ND * sizeof(float);
    float*  H    = (float*)ws;   ws += (size_t)NB * ND * sizeof(float);
    bf16_t* Hbf  = (bf16_t*)ws;

    // 1) pack concat([mean,var]) -> bf16 (NB, 32768)
    pack_x_bf16<<<(NB * NF * ND) / 256, 256, 0, stream>>>(mean, varr, Xbf);

    // 2) W1 (32768x256) -> W1T (256x32768) ; W2 (256x8192) -> W2T (8192x256)
    transpose_bf16<<<dim3(ND / 32, IN_DIM / 32), dim3(32, 8), 0, stream>>>(
        W1, W1T, IN_DIM, ND);
    transpose_bf16<<<dim3(OUT_DIM / 32, ND / 32), dim3(32, 8), 0, stream>>>(
        W2, W2T, ND, OUT_DIM);

    // 3) GEMM1, split-K=8 -> f32 partials (512 blocks)
    wmma_gemm_bf16<<<dim3(ND / 64, NB / 64, KSPLIT), 128, 0, stream>>>(
        Xbf, W1T, nullptr, Part, NB, ND, IN_DIM, IN_DIM / KSPLIT, 0);

    // 4) fixed-order split-K reduce + bias + exact GELU -> H
    reduce_bias_gelu<<<(NB * ND) / 256, 256, 0, stream>>>(Part, b1, H);

    // 5) LayerNorm + kWTA -> Hbf (bf16)
    ln_kwta<<<NB, ND, 0, stream>>>(H, gamma, beta, Hbf);

    // 6) GEMM2 + bias + tanh -> out (2048 blocks)
    wmma_gemm_bf16<<<dim3(OUT_DIM / 64, NB / 64, 1), 128, 0, stream>>>(
        Hbf, W2T, b2, out, NB, OUT_DIM, ND, ND, 1);
}
